// BaseAttention_49529562858124
// MI455X (gfx1250) — matrix operand order repaired and checked by reading, not compiled, not run on hardware
//
#include <hip/hip_runtime.h>

typedef __attribute__((ext_vector_type(16))) __bf16 v16bf;
typedef __attribute__((ext_vector_type(4)))  __bf16 v4bf;
typedef __attribute__((ext_vector_type(4)))  float  v4f;
typedef __attribute__((ext_vector_type(8)))  float  v8f;

#define S_DIM 2048
#define D_DIM 128
#define H_DIM 16
#define BLK_M 64
#define BLK_N 64
#define NITER (S_DIM / BLK_N)
#define MASK_VAL -1e9f

__device__ __forceinline__ v4bf cvt4(float4 f) {
    v4f t = {f.x, f.y, f.z, f.w};
    return __builtin_convertvector(t, v4bf);   // hardware fptrunc f32->bf16 (RNE)
}

union V16 {
    v4bf  q[4];
    uint4 u[2];
    v16bf v;
};

__global__ __launch_bounds__(128, 1)
void fa_fwd_bf16_wmma(const float* __restrict__ Q,
                      const float* __restrict__ K,
                      const float* __restrict__ V,
                      const unsigned char* __restrict__ M,
                      float* __restrict__ O) {
    const int b    = blockIdx.z;
    const int h    = blockIdx.y;
    const int tid  = threadIdx.x;
    const int w    = tid >> 5;          // wave id 0..3
    const int lane = tid & 31;
    const int l15  = lane & 15;
    const int lh   = (lane >> 4) & 1;   // lane half-group

    const size_t bh = ((size_t)b * H_DIM + h) * (size_t)S_DIM * D_DIM;
    const float* Qg = Q + bh;
    const float* Kg = K + bh;
    const float* Vg = V + bh;
    float*       Og = O + bh;
    const unsigned char* Mg = M + (size_t)b * S_DIM * S_DIM;

    const int qBlockBase = blockIdx.x * BLK_M;
    const int qBase      = qBlockBase + w * 16;

    __shared__ __align__(32) __bf16 Kt[BLK_N * D_DIM];        // 64x128, row-major
    __shared__ __align__(32) __bf16 Vt[D_DIM * BLK_N];        // 128x64, V transposed
    __shared__ __align__(32) __bf16 Pw[4][16 * BLK_N];        // per-wave P tile
    __shared__ __align__(16) unsigned char Mt[BLK_M * BLK_N]; // 64x64 mask tile

    // ---- Q A-fragments (16 rows x 128 cols -> 4 k-steps of 32), bf16 in regs ----
    v16bf aq[4];
    {
        const float* qrow = Qg + (size_t)(qBase + l15) * D_DIM;
        #pragma unroll
        for (int ks = 0; ks < 4; ++ks) {
            const int cb = ks * 32 + lh * 8;
            V16 a;
            a.q[0] = cvt4(*(const float4*)(qrow + cb));
            a.q[1] = cvt4(*(const float4*)(qrow + cb + 4));
            a.q[2] = cvt4(*(const float4*)(qrow + cb + 16));
            a.q[3] = cvt4(*(const float4*)(qrow + cb + 20));
            aq[ks] = a.v;
        }
    }

    v8f Oacc[8];
    #pragma unroll
    for (int i = 0; i < 8; ++i) Oacc[i] = (v8f){0,0,0,0,0,0,0,0};
    v8f mrow = (v8f){-1e30f,-1e30f,-1e30f,-1e30f,-1e30f,-1e30f,-1e30f,-1e30f};
    v8f lrow = (v8f){0,0,0,0,0,0,0,0};

    const float scale = 0.08838834764831845f;   // 1/sqrt(128)

    for (int it = 0; it < NITER; ++it) {
        const int kvBase = it * BLK_N;

        if (it + 1 < NITER) {   // speculative prefetch of next tiles -> global_prefetch_b8
            __builtin_prefetch(&Kg[(size_t)(kvBase + BLK_N) * D_DIM + tid * 16], 0, 1);
            __builtin_prefetch(&Vg[(size_t)(kvBase + BLK_N) * D_DIM + tid * 16], 0, 1);
        }

        __syncthreads();
        // ---- mask tile: 2 coalesced b128 loads/thread -> LDS ----
        #pragma unroll
        for (int i = 0; i < 2; ++i) {
            const int idx  = i * 128 + tid;        // 0..255 (uint4 units)
            const int mr   = idx >> 2;             // 0..63 rows
            const int mc   = (idx & 3) * 16;       // 0..48 col base
            *(uint4*)&Mt[mr * BLK_N + mc] =
                *(const uint4*)&Mg[(size_t)(qBlockBase + mr) * S_DIM + kvBase + mc];
        }
        // ---- K tile: fp32 -> bf16, row-major, b64 stores ----
        #pragma unroll
        for (int i = 0; i < 16; ++i) {
            const int idx = i * 128 + tid;         // float4 index within 64x32 float4 tile
            const int row = idx >> 5;
            const int cg  = idx & 31;
            v4bf kb = cvt4(*(const float4*)&Kg[(size_t)(kvBase + row) * D_DIM + cg * 4]);
            *(v4bf*)&Kt[row * D_DIM + cg * 4] = kb;
        }
        // ---- V tile: fp32 -> bf16 transposed; pack 2 kv rows -> b32 stores ----
        #pragma unroll
        for (int i = 0; i < 8; ++i) {
            const int idx = i * 128 + tid;         // 0..1023 (row-pair x float4-col)
            const int rp  = idx >> 5;              // kv row pair 0..31
            const int cg  = idx & 31;
            v4bf va = cvt4(*(const float4*)&Vg[(size_t)(kvBase + rp * 2) * D_DIM + cg * 4]);
            v4bf vb = cvt4(*(const float4*)&Vg[(size_t)(kvBase + rp * 2 + 1) * D_DIM + cg * 4]);
            #pragma unroll
            for (int d2 = 0; d2 < 4; ++d2) {
                union { __bf16 h[2]; unsigned u; } p;
                p.h[0] = va[d2];
                p.h[1] = vb[d2];
                *(unsigned*)&Vt[(cg * 4 + d2) * BLK_N + rp * 2] = p.u;
            }
        }
        __syncthreads();

        // ---- S = Q K^T : per N-tile, batch all 4 B-fragments then chain 4 WMMAs ----
        v8f sf[4];
        #pragma unroll
        for (int t = 0; t < 4; ++t) {
            V16 bb[4];
            #pragma unroll
            for (int ks = 0; ks < 4; ++ks) {
                const int off = (t * 16 + l15) * D_DIM + ks * 32 + lh * 8;
                bb[ks].u[0] = *(const uint4*)&Kt[off];
                bb[ks].u[1] = *(const uint4*)&Kt[off + 16];
            }
            v8f s = (v8f){0,0,0,0,0,0,0,0};
            #pragma unroll
            for (int ks = 0; ks < 4; ++ks)
                s = __builtin_amdgcn_wmma_f32_16x16x32_bf16(
                        false, aq[ks], false, bb[ks].v, (short)0, s, false, false);
            sf[t] = s;
        }

        // ---- scale + additive mask (bytes from LDS tile) ----
        #pragma unroll
        for (int t = 0; t < 4; ++t) {
            #pragma unroll
            for (int j = 0; j < 8; ++j) {
                const unsigned char mb =
                    Mt[(w * 16 + j + lh * 8) * BLK_N + t * 16 + l15];
                sf[t][j] = sf[t][j] * scale + (mb ? MASK_VAL : 0.0f);
            }
        }

        // ---- streaming softmax (row reductions across the 16-lane half-groups) ----
        v8f alpha;
        #pragma unroll
        for (int j = 0; j < 8; ++j) {
            float r = fmaxf(fmaxf(sf[0][j], sf[1][j]), fmaxf(sf[2][j], sf[3][j]));
            r = fmaxf(r, __shfl_xor(r, 1, 32));
            r = fmaxf(r, __shfl_xor(r, 2, 32));
            r = fmaxf(r, __shfl_xor(r, 4, 32));
            r = fmaxf(r, __shfl_xor(r, 8, 32));
            const float mnew = fmaxf(mrow[j], r);
            const float al   = __expf(mrow[j] - mnew);
            mrow[j] = mnew;
            float rs = 0.0f;
            #pragma unroll
            for (int t = 0; t < 4; ++t) {
                const float p = __expf(sf[t][j] - mnew);
                sf[t][j] = p;
                rs += p;
            }
            rs += __shfl_xor(rs, 1, 32);
            rs += __shfl_xor(rs, 2, 32);
            rs += __shfl_xor(rs, 4, 32);
            rs += __shfl_xor(rs, 8, 32);
            lrow[j] = lrow[j] * al + rs;
            alpha[j] = al;
        }
        #pragma unroll
        for (int nt = 0; nt < 8; ++nt)
            #pragma unroll
            for (int j = 0; j < 8; ++j)
                Oacc[nt][j] *= alpha[j];

        // ---- P: C-layout -> A-layout via per-wave LDS region ----
        #pragma unroll
        for (int t = 0; t < 4; ++t)
            #pragma unroll
            for (int j = 0; j < 8; ++j)
                Pw[w][(j + lh * 8) * BLK_N + t * 16 + l15] = (__bf16)sf[t][j];
        asm volatile("s_wait_dscnt 0" ::: "memory");

        V16 ap[2];
        #pragma unroll
        for (int ks = 0; ks < 2; ++ks) {
            const int base = l15 * BLK_N + ks * 32 + lh * 16;
            ap[ks].u[0] = *(const uint4*)&Pw[w][base];
            ap[ks].u[1] = *(const uint4*)&Pw[w][base + 8];
        }

        // ---- O += P V : 8 D-tiles x 2 k-steps; batch both fragments per tile ----
        #pragma unroll
        for (int nt = 0; nt < 8; ++nt) {
            V16 b0, b1;
            {
                const int off = (nt * 16 + l15) * BLK_N;
                b0.u[0] = *(const uint4*)&Vt[off + lh * 16];
                b0.u[1] = *(const uint4*)&Vt[off + lh * 16 + 8];
                b1.u[0] = *(const uint4*)&Vt[off + 32 + lh * 16];
                b1.u[1] = *(const uint4*)&Vt[off + 32 + lh * 16 + 8];
            }
            Oacc[nt] = __builtin_amdgcn_wmma_f32_16x16x32_bf16(
                           false, ap[0].v, false, b0.v, (short)0, Oacc[nt], false, false);
            Oacc[nt] = __builtin_amdgcn_wmma_f32_16x16x32_bf16(
                           false, ap[1].v, false, b1.v, (short)0, Oacc[nt], false, false);
        }
    }

    // ---- epilogue: O / l ----
    v8f invl;
    #pragma unroll
    for (int j = 0; j < 8; ++j) invl[j] = 1.0f / lrow[j];
    #pragma unroll
    for (int nt = 0; nt < 8; ++nt) {
        #pragma unroll
        for (int j = 0; j < 8; ++j) {
            const int qr = qBase + j + lh * 8;
            const int dc = nt * 16 + l15;
            Og[(size_t)qr * D_DIM + dc] = Oacc[nt][j] * invl[j];
        }
    }
}

extern "C" void kernel_launch(void* const* d_in, const int* in_sizes, int n_in,
                              void* d_out, int out_size, void* d_ws, size_t ws_size,
                              hipStream_t stream) {
    (void)in_sizes; (void)n_in; (void)out_size; (void)d_ws; (void)ws_size;
    const float* q = (const float*)d_in[0];
    const float* k = (const float*)d_in[1];
    const float* v = (const float*)d_in[2];
    const unsigned char* m = (const unsigned char*)d_in[3];
    float* out = (float*)d_out;

    dim3 grid(S_DIM / BLK_M, H_DIM, 4);
    dim3 block(128);
    fa_fwd_bf16_wmma<<<grid, block, 0, stream>>>(q, k, v, m, out);
}
